// RePaIcoConvLayer_16664473108744
// MI455X (gfx1250) — compile-verified, matrix operand order checked by reading, not run on hardware
//
#include <hip/hip_runtime.h>
#include <hip/hip_bf16.h>

// ---------------------------------------------------------------------------
// RePaIcoConvLayer for MI455X (gfx1250), wave32, WMMA bf16 path.
//
// out[b, o, n] = sum_{c,k} feat[b,n, c*9+k] * W[o, c*9+k] + bias[o]
// feat[b,n, c*9+k] = sum_{j=0..2} x[b, c, idx[n, k*3+j]] * w[n, k*3+j]
// ---------------------------------------------------------------------------

#define N_VERT   40962
#define NEIGH    9
#define K3       27          // NEIGH * 3
#define IN_F     64
#define OUT_F    64
#define BATCH    4
#define KDIM     576         // IN_F * NEIGH
#define TOT_ROWS (BATCH * N_VERT)   // 163848
#define M_TILE   32
#define KSTEPS   (KDIM / 32)        // 18

typedef __attribute__((ext_vector_type(16))) __bf16 v16bf;
typedef __attribute__((ext_vector_type(8)))  float  v8f;

union Frag16 {
    v16bf v;
    uint4 q[2];
};

__device__ __forceinline__ unsigned short f32_to_bf16_rne(float f) {
    union { float f; unsigned int u; } cvt;
    cvt.f = f;
    unsigned int u = cvt.u;
    u += 0x7FFFu + ((u >> 16) & 1u);   // round to nearest even
    return (unsigned short)(u >> 16);
}

// --------------------------- pre-pass: W -> bf16 ---------------------------
__global__ void convert_w_bf16(const float* __restrict__ W,
                               unsigned short* __restrict__ Wb, int n) {
    int i = blockIdx.x * blockDim.x + threadIdx.x;
    if (i < n) Wb[i] = f32_to_bf16_rne(W[i]);
}

// ------------------------------ fused kernel -------------------------------
__global__ __launch_bounds__(256)
void repa_ico_conv_kernel(const float* __restrict__ x,
                          const int*   __restrict__ nidx,
                          const float* __restrict__ nwgt,
                          const unsigned short* __restrict__ Wb,   // bf16 W [64][576]
                          const float* __restrict__ bias,
                          float* __restrict__ out) {
    __shared__ __align__(16) unsigned short sFeat[M_TILE * KDIM]; // 36 KB bf16 feat tile
    __shared__ int   sIdx[M_TILE * K3];                           // per-row gather indices
    __shared__ float sWgt[M_TILE * K3];                           // per-row gather weights
    __shared__ float sOut[M_TILE * OUT_F];                        // 8 KB result tile

    const int tid   = threadIdx.x;
    const int wgRow = blockIdx.x * M_TILE;

    // ---- stage 0: stage indices + weights for the 32 rows of this tile ----
    for (int i = tid; i < M_TILE * K3; i += 256) {
        const int m = i / K3;
        const int j = i - m * K3;
        const int r = wgRow + m;
        if (r < TOT_ROWS) {
            const int n = r % N_VERT;
            sIdx[i] = nidx[n * K3 + j];
            sWgt[i] = nwgt[n * K3 + j];
        } else {
            sIdx[i] = 0;
            sWgt[i] = 0.0f;
        }
    }
    __syncthreads();

    // ---- stage A: gather + 3-term weighted sum -> bf16 feat tile in LDS ----
    // feat[m][c*9+k] ; 32*576 = 18432 elements, 72 per thread.
    for (int e = tid; e < M_TILE * KDIM; e += 256) {
        const int m   = e / KDIM;
        const int col = e - m * KDIM;          // c*9 + kk
        const int c   = col / NEIGH;
        const int kk  = col - c * NEIGH;
        const int r   = wgRow + m;
        float acc = 0.0f;
        if (r < TOT_ROWS) {
            const int bb = r / N_VERT;
            const int base = m * K3 + kk * 3;
            const float* __restrict__ xb = x + (size_t)(bb * IN_F + c) * N_VERT;
            acc  = xb[sIdx[base + 0]] * sWgt[base + 0];
            acc += xb[sIdx[base + 1]] * sWgt[base + 1];
            acc += xb[sIdx[base + 2]] * sWgt[base + 2];
        }
        sFeat[e] = f32_to_bf16_rne(acc);
    }
    __syncthreads();

    // ---- stage B: 8 waves x one 16x16 tile, K = 576 in 18 WMMA steps ------
    const int wave  = tid >> 5;
    const int lane  = tid & 31;
    const int lmod  = lane & 15;       // M (for A) / N (for B) within tile
    const int lhi   = lane >> 4;       // K half selector
    const int mbase = (wave >> 2) * 16;       // 0 or 16
    const int obase = (wave & 3) * 16;        // 0,16,32,48

    const unsigned short* __restrict__ arow = sFeat + (mbase + lmod) * KDIM;
    const unsigned short* __restrict__ brow = Wb + (size_t)(obase + lmod) * KDIM;

    v8f acc = {};
#pragma unroll
    for (int ks = 0; ks < KSTEPS; ++ks) {
        const int k0 = ks * 32;
        Frag16 a, b;
        // 16-bit A layout: lane = M, VGPR0-3 hold K = lhi*8 .. +7,
        // VGPR4-7 hold K = 16 + lhi*8 .. +7 (two 16B chunks).
        a.q[0] = *(const uint4*)(arow + k0 + lhi * 8);
        a.q[1] = *(const uint4*)(arow + k0 + 16 + lhi * 8);
        // B (32x16) symmetric layout: lane = N, same K chunking.
        b.q[0] = *(const uint4*)(brow + k0 + lhi * 8);
        b.q[1] = *(const uint4*)(brow + k0 + 16 + lhi * 8);
        acc = __builtin_amdgcn_wmma_f32_16x16x32_bf16(
            /*neg_a=*/false, a.v, /*neg_b=*/false, b.v,
            /*c_mod=*/(short)0, acc, /*reuse_a=*/false, /*reuse_b=*/false);
    }

    // ---- stage C: bias add, stash tile in LDS for coalesced store ---------
    const float bv = bias[obase + lmod];
#pragma unroll
    for (int rr = 0; rr < 8; ++rr) {
        const int m = mbase + lhi * 8 + rr;    // C/D layout: M = vgpr + lhi*8
        sOut[m * OUT_F + obase + lmod] = acc[rr] + bv;
    }
    __syncthreads();

    // out is (B, 64, V): contiguous along vertex index -> iterate o-major so
    // consecutive lanes write consecutive vertices (128B segments).
    for (int i = tid; i < M_TILE * OUT_F; i += 256) {
        const int o = i >> 5;          // i / 32
        const int m = i & 31;          // i % 32
        const int r = wgRow + m;
        if (r < TOT_ROWS) {
            const int bb = r / N_VERT;
            const int n  = r % N_VERT;
            out[((size_t)(bb * OUT_F + o)) * N_VERT + n] = sOut[m * OUT_F + o];
        }
    }
}

// ------------------------------- launcher ----------------------------------
extern "C" void kernel_launch(void* const* d_in, const int* in_sizes, int n_in,
                              void* d_out, int out_size, void* d_ws, size_t ws_size,
                              hipStream_t stream) {
    const float* x    = (const float*)d_in[0];
    const int*   nidx = (const int*)  d_in[1];
    const float* nwgt = (const float*)d_in[2];
    const float* W    = (const float*)d_in[3];
    const float* bias = (const float*)d_in[4];
    float* out = (float*)d_out;

    unsigned short* Wb = (unsigned short*)d_ws;   // 64*576 bf16 = 73728 B

    const int wElems = OUT_F * KDIM;              // 36864
    convert_w_bf16<<<(wElems + 255) / 256, 256, 0, stream>>>(W, Wb, wElems);

    const int nBlocks = (TOT_ROWS + M_TILE - 1) / M_TILE;   // 5121
    repa_ico_conv_kernel<<<nBlocks, 256, 0, stream>>>(x, nidx, nwgt, Wb, bias, out);
}
